// System2Reasoner_36670430773781
// MI455X (gfx1250) — matrix operand order repaired and checked by reading, not compiled, not run on hardware
//
#include <hip/hip_runtime.h>
#include <hip/hip_bf16.h>

typedef __bf16 bf16_t;
typedef __attribute__((ext_vector_type(16))) bf16_t v16bf;
typedef __attribute__((ext_vector_type(8)))  bf16_t v8bf;
typedef __attribute__((ext_vector_type(8)))  float  v8f;
typedef unsigned short u16;
typedef unsigned int   u32;

#define NPATCH 2048
#define NNODES 100000
#define DIM    256
#define TOPK   50
#define KPAD   64
#define NSEG   8
#define SEGCNT (NNODES / NSEG)      // 12500
#define MROWS  64                   // patches per workgroup
#define NCOLS  32                   // nodes per workgroup iteration
#define TAU    0.02f
#define NEGBIG (-3.0e38f)

__device__ __forceinline__ u16 f32_to_bf16_rne(float f) {
  u32 u = __float_as_uint(f);
  u32 lsb = (u >> 16) & 1u;
  u += 0x7fffu + lsb;
  return (u16)(u >> 16);
}
__device__ __forceinline__ float bf16u_to_f32(u16 h) {
  return __uint_as_float(((u32)h) << 16);
}

// ---------------------------------------------------------------------------
// Kernel 1: split f32 -> bf16 hi + bf16 lo (x ~= hi + lo), vectorized x4
// ---------------------------------------------------------------------------
__global__ __launch_bounds__(256)
void split_bf16_kernel(const float* __restrict__ in, u16* __restrict__ hi,
                       u16* __restrict__ lo, int n4) {
  int i = blockIdx.x * 256 + threadIdx.x;
  if (i >= n4) return;
  float4 f = ((const float4*)in)[i];
  u16 h0 = f32_to_bf16_rne(f.x);
  u16 h1 = f32_to_bf16_rne(f.y);
  u16 h2 = f32_to_bf16_rne(f.z);
  u16 h3 = f32_to_bf16_rne(f.w);
  ushort4 H; H.x = h0; H.y = h1; H.z = h2; H.w = h3;
  ushort4 L;
  L.x = f32_to_bf16_rne(f.x - bf16u_to_f32(h0));
  L.y = f32_to_bf16_rne(f.y - bf16u_to_f32(h1));
  L.z = f32_to_bf16_rne(f.z - bf16u_to_f32(h2));
  L.w = f32_to_bf16_rne(f.w - bf16u_to_f32(h3));
  ((ushort4*)hi)[i] = H;
  ((ushort4*)lo)[i] = L;
}

// ---------------------------------------------------------------------------
// Kernel 2: fused split-bf16 WMMA GEMM + streaming per-row top-50.
// Grid: (NPATCH/64, NSEG), 256 threads (8 waves).
// Wave w = (msub = w&3, nsub = w>>2): computes C tile for patches
// [mtile*64 + msub*16 ..+16) x nodes [node_base + nsub*16 ..+16).
// B tiles are shared by 4 waves (identical addresses -> WGP$ broadcast),
// cutting L2 traffic 4x vs per-wave B. A tile held in VGPRs for whole kernel.
// The three split-precision terms accumulate into three INDEPENDENT
// accumulators (3-way ILP in the matrix pipe; merged with 16 f32 adds).
// ---------------------------------------------------------------------------
__global__ __launch_bounds__(256)
void sim_topk_kernel(const u16* __restrict__ ph, const u16* __restrict__ pl,
                     const u16* __restrict__ nh, const u16* __restrict__ nl,
                     float* __restrict__ seg_vals, int* __restrict__ seg_idx) {
  __shared__ float s_sim[MROWS][36];       // 32 cols padded to 36 (bank-safe)
  __shared__ float s_val[MROWS][KPAD];
  __shared__ int   s_idx[MROWS][KPAD];
  __shared__ u32   s_flag[MROWS];
  __shared__ float s_min[MROWS];
  __shared__ int   s_minpos[MROWS];

  const int tid   = threadIdx.x;
  const int wave  = tid >> 5;
  const int lane  = tid & 31;
  const int l16   = lane & 15;
  const int msub  = wave & 3;
  const int nsub  = wave >> 2;
  const int mtile = blockIdx.x;
  const int seg   = blockIdx.y;
  const int seg_base = seg * SEGCNT;
  const int seg_end  = seg_base + SEGCNT;

  for (int i = tid; i < MROWS * KPAD; i += 256) {
    (&s_val[0][0])[i] = NEGBIG;
    (&s_idx[0][0])[i] = 0;
  }
  if (tid < MROWS) { s_min[tid] = NEGBIG; s_minpos[tid] = 0; s_flag[tid] = 0u; }

  // Preload A tile (16 patches x 256 dims), hi & lo, into registers.
  // 16-bit A 16x32 layout: lanes 0-15: M=lane, K = {32c..+7, 32c+16..+23};
  // lanes 16-31: same M, K shifted by +8.
  const int  arow = mtile * MROWS + msub * 16 + l16;
  const int  aoff = (lane < 16) ? 0 : 8;
  const u16* prh  = ph + (size_t)arow * DIM;
  const u16* prl  = pl + (size_t)arow * DIM;
  v16bf Ahi[8], Alo[8];
#pragma unroll
  for (int c = 0; c < 8; ++c) {
    const int k0 = c * 32 + aoff;
    v8bf h0 = *(const v8bf*)(prh + k0);
    v8bf h1 = *(const v8bf*)(prh + k0 + 16);
    v8bf l0 = *(const v8bf*)(prl + k0);
    v8bf l1 = *(const v8bf*)(prl + k0 + 16);
    Ahi[c] = __builtin_shufflevector(h0, h1, 0,1,2,3,4,5,6,7,8,9,10,11,12,13,14,15);
    Alo[c] = __builtin_shufflevector(l0, l1, 0,1,2,3,4,5,6,7,8,9,10,11,12,13,14,15);
  }
  __syncthreads();

  const int iters = (SEGCNT + NCOLS - 1) / NCOLS;
  for (int it = 0; it < iters; ++it) {
    const int node_base = seg_base + it * NCOLS;
    int brow = node_base + nsub * 16 + l16;
    if (brow > NNODES - 1) brow = NNODES - 1;   // clamp; invalid cols skipped later
    // 16-bit B 32x16 layout: lanes 0-15: N=lane, K = {32c..+7, 32c+16..+23};
    // lanes 16-31: K shifted by +8. B[k][n] = nodes[n][k] -> row-contiguous.
    const int  boff = (lane < 16) ? 0 : 8;
    const u16* nrh  = nh + (size_t)brow * DIM;
    const u16* nrl  = nl + (size_t)brow * DIM;

    // Prefetch node rows 2 tiles ahead (covers 32 rows x 512B x {hi,lo};
    // each lane of waves 0-1 pulls 2 cachelines per array). Counter-free.
    if (wave < 2) {
      const int prow = node_base + 2 * NCOLS + wave * 16 + (lane >> 1);
      if (prow < seg_end) {
        const u16* a = nh + (size_t)prow * DIM + (lane & 1) * 64;
        const u16* b = nl + (size_t)prow * DIM + (lane & 1) * 64;
        __builtin_prefetch(a, 0, 3);
        __builtin_prefetch(a + 128, 0, 3);
        __builtin_prefetch(b, 0, 3);
        __builtin_prefetch(b + 128, 0, 3);
      }
    }

    v8f acc1 = {};   // hi*hi
    v8f acc2 = {};   // hi*lo
    v8f acc3 = {};   // lo*hi
#pragma unroll
    for (int c = 0; c < 8; ++c) {
      const int k0 = c * 32 + boff;
      v8bf bh0 = *(const v8bf*)(nrh + k0);
      v8bf bh1 = *(const v8bf*)(nrh + k0 + 16);
      v8bf bl0 = *(const v8bf*)(nrl + k0);
      v8bf bl1 = *(const v8bf*)(nrl + k0 + 16);
      v16bf Bhi = __builtin_shufflevector(bh0, bh1, 0,1,2,3,4,5,6,7,8,9,10,11,12,13,14,15);
      v16bf Blo = __builtin_shufflevector(bl0, bl1, 0,1,2,3,4,5,6,7,8,9,10,11,12,13,14,15);
      // split-precision: hi*hi + hi*lo + lo*hi  (~f32 accuracy),
      // three independent accumulation chains for matrix-pipe ILP.
      acc1 = __builtin_amdgcn_wmma_f32_16x16x32_bf16(false, Ahi[c], false, Bhi, (short)0, acc1, false, false);
      acc2 = __builtin_amdgcn_wmma_f32_16x16x32_bf16(false, Ahi[c], false, Blo, (short)0, acc2, false, false);
      acc3 = __builtin_amdgcn_wmma_f32_16x16x32_bf16(false, Alo[c], false, Bhi, (short)0, acc3, false, false);
    }
    const v8f acc = (acc2 + acc3) + acc1;   // 16 VALU f32 adds

    // Scatter C tile: lane l, vgpr v -> row msub*16 + v + (l<16?0:8),
    //                                  col nsub*16 + l%16
    const int m0   = msub * 16 + ((lane < 16) ? 0 : 8);
    const int ncol = nsub * 16 + l16;
#pragma unroll
    for (int v = 0; v < 8; ++v) s_sim[m0 + v][ncol] = acc[v];
    __syncthreads();

    // Parallel threshold scan: 4 threads/row x 8 candidates each.
    {
      const int   row = tid >> 2;
      const int   c0  = (tid & 3) * 8;
      const float thr = s_min[row];
      u32 bits = 0u;
#pragma unroll
      for (int j = 0; j < 8; ++j) {
        const int c = c0 + j;
        if ((node_base + c) < seg_end && s_sim[row][c] > thr)
          bits |= (1u << c);
      }
      if (bits) atomicOr(&s_flag[row], bits);
    }
    __syncthreads();

    // Rare serial insert: one leader per row replaces current min, rescans 50.
    if (tid < MROWS) {
      const int row  = tid;
      float curmin   = s_min[row];
      int   minpos   = s_minpos[row];
      u32 bits = s_flag[row];
      s_flag[row] = 0u;
      while (bits) {
        const int b = __builtin_ctz(bits);
        bits &= bits - 1u;
        const float v = s_sim[row][b];
        if (v > curmin) {
          s_val[row][minpos] = v;
          s_idx[row][minpos] = node_base + b;
          curmin = s_val[row][0]; minpos = 0;
          for (int j = 1; j < TOPK; ++j) {
            const float vj = s_val[row][j];
            if (vj < curmin) { curmin = vj; minpos = j; }
          }
        }
      }
      s_min[row] = curmin; s_minpos[row] = minpos;
    }
    __syncthreads();
  }

  // Emit (unsorted) per-segment top-50 (padded to 64 with NEGBIG).
  {
    const int    row  = tid >> 2;
    const int    j0   = (tid & 3) * 16;
    const size_t base = ((size_t)(mtile * MROWS + row) * NSEG + seg) * KPAD;
#pragma unroll
    for (int j = 0; j < 16; ++j) {
      seg_vals[base + j0 + j] = s_val[row][j0 + j];
      seg_idx [base + j0 + j] = s_idx [row][j0 + j];
    }
  }
}

// ---------------------------------------------------------------------------
// Kernel 3: per patch — merge 8x64 candidates (exact rank selection),
// softmax(tau), gather f32 node rows, message, residual + l2norm.
// One block per patch, 256 threads (2 candidates/thread; thread = dim).
// ---------------------------------------------------------------------------
__global__ __launch_bounds__(256)
void merge_message_kernel(const float* __restrict__ patches,
                          const float* __restrict__ nodes,
                          const float* __restrict__ seg_vals,
                          const int*   __restrict__ seg_idx,
                          float* __restrict__ out,
                          float* __restrict__ evidence) {
  __shared__ float c_val[NSEG * KPAD];   // 512
  __shared__ int   c_idx[NSEG * KPAD];
  __shared__ float sel_v[TOPK];
  __shared__ int   sel_i[TOPK];
  __shared__ float red[256];
  __shared__ float s_max, s_sum;

  const int tid = threadIdx.x;
  const int p   = blockIdx.x;
  const size_t cbase = (size_t)p * NSEG * KPAD;
  float2 mv = ((const float2*)(seg_vals + cbase))[tid];
  int2   mi = ((const int2*)(seg_idx + cbase))[tid];
  ((float2*)c_val)[tid] = mv;
  ((int2*)c_idx)[tid]   = mi;
  __syncthreads();

  // Exact ranks (ties broken by slot index -> deterministic).
  const int s0 = tid * 2;
  int r0 = 0, r1 = 0;
  for (int j = 0; j < NSEG * KPAD; ++j) {
    const float vj = c_val[j];
    r0 += ((vj > mv.x) || (vj == mv.x && j < s0)) ? 1 : 0;
    r1 += ((vj > mv.y) || (vj == mv.y && j < s0 + 1)) ? 1 : 0;
  }
  if (r0 == 0) s_max = mv.x;
  if (r1 == 0) s_max = mv.y;
  if (r0 < TOPK) { sel_v[r0] = mv.x; sel_i[r0] = mi.x; }
  if (r1 < TOPK) { sel_v[r1] = mv.y; sel_i[r1] = mi.y; }
  __syncthreads();

  if (tid == 0) {
    const float m = s_max;
    float s = 0.f;
    for (int k = 0; k < TOPK; ++k) {
      const float e = __expf((sel_v[k] - m) * (1.0f / TAU));
      sel_v[k] = e; s += e;
    }
    s_sum = s;
    evidence[p] = m;
  }
  __syncthreads();

  const int d = tid;           // DIM == blockDim
  float acc = 0.f;
  for (int k = 0; k < TOPK; ++k)
    acc += sel_v[k] * nodes[(size_t)sel_i[k] * DIM + d];
  acc = patches[(size_t)p * DIM + d] + acc / s_sum;

  red[tid] = acc * acc;
  __syncthreads();
  for (int s = 128; s > 0; s >>= 1) { if (tid < s) red[tid] += red[tid + s]; __syncthreads(); }
  const float inv = 1.0f / fmaxf(sqrtf(red[0]), 1e-12f);
  out[DIM + (size_t)p * DIM + d] = acc * inv;
}

// ---------------------------------------------------------------------------
// Kernel 4: global feature — softmax over evidence, weighted sum of updated,
// l2norm -> out[0:256]. Single block, 256 threads (thread = dim).
// ---------------------------------------------------------------------------
__global__ __launch_bounds__(256)
void global_feature_kernel(const float* __restrict__ evidence,
                           const float* __restrict__ updated,
                           float* __restrict__ out) {
  __shared__ float red[256];
  __shared__ float w[NPATCH];
  const int tid = threadIdx.x;

  float m = NEGBIG;
  for (int i = tid; i < NPATCH; i += 256) m = fmaxf(m, evidence[i]);
  red[tid] = m; __syncthreads();
  for (int s = 128; s > 0; s >>= 1) { if (tid < s) red[tid] = fmaxf(red[tid], red[tid + s]); __syncthreads(); }
  const float mx = red[0];
  __syncthreads();

  float sum = 0.f;
  for (int i = tid; i < NPATCH; i += 256) {
    const float e = __expf((evidence[i] - mx) * (1.0f / TAU));
    w[i] = e; sum += e;
  }
  red[tid] = sum; __syncthreads();
  for (int s = 128; s > 0; s >>= 1) { if (tid < s) red[tid] += red[tid + s]; __syncthreads(); }
  const float inv_sum = 1.0f / red[0];
  __syncthreads();

  float acc = 0.f;
  for (int p = 0; p < NPATCH; ++p)
    acc += updated[(size_t)p * DIM + tid] * w[p];
  acc *= inv_sum;

  red[tid] = acc * acc; __syncthreads();
  for (int s = 128; s > 0; s >>= 1) { if (tid < s) red[tid] += red[tid + s]; __syncthreads(); }
  out[tid] = acc / fmaxf(sqrtf(red[0]), 1e-12f);
}

// ---------------------------------------------------------------------------
extern "C" void kernel_launch(void* const* d_in, const int* in_sizes, int n_in,
                              void* d_out, int out_size, void* d_ws, size_t ws_size,
                              hipStream_t stream) {
  (void)in_sizes; (void)n_in; (void)out_size; (void)ws_size;
  const float* patches = (const float*)d_in[0];   // 2048 x 256 f32
  const float* nodes   = (const float*)d_in[1];   // 100000 x 256 f32
  float*       out     = (float*)d_out;           // [256 global | 2048x256 updated]

  char* ws = (char*)d_ws;
  size_t o = 0;
  u16* ph = (u16*)(ws + o); o += (size_t)NPATCH * DIM * sizeof(u16);
  u16* pl = (u16*)(ws + o); o += (size_t)NPATCH * DIM * sizeof(u16);
  u16* nh = (u16*)(ws + o); o += (size_t)NNODES * DIM * sizeof(u16);
  u16* nl = (u16*)(ws + o); o += (size_t)NNODES * DIM * sizeof(u16);
  float* seg_vals = (float*)(ws + o); o += (size_t)NPATCH * NSEG * KPAD * sizeof(float);
  int*   seg_idx  = (int*)  (ws + o); o += (size_t)NPATCH * NSEG * KPAD * sizeof(int);
  float* evidence = (float*)(ws + o); o += (size_t)NPATCH * sizeof(float);

  const int pn4 = NPATCH * DIM / 4;
  const int nn4 = NNODES * DIM / 4;
  split_bf16_kernel<<<(pn4 + 255) / 256, 256, 0, stream>>>(patches, ph, pl, pn4);
  split_bf16_kernel<<<(nn4 + 255) / 256, 256, 0, stream>>>(nodes, nh, nl, nn4);

  sim_topk_kernel<<<dim3(NPATCH / MROWS, NSEG), 256, 0, stream>>>(ph, pl, nh, nl, seg_vals, seg_idx);

  merge_message_kernel<<<NPATCH, 256, 0, stream>>>(patches, nodes, seg_vals, seg_idx, out, evidence);

  global_feature_kernel<<<1, 256, 0, stream>>>(evidence, out + DIM, out);
}
